// Attention_Module_26371099198420
// MI455X (gfx1250) — matrix operand order repaired and checked by reading, not compiled, not run on hardware
//
#include <hip/hip_runtime.h>

// ---------------------------------------------------------------------------
// Problem constants (from reference)
// ---------------------------------------------------------------------------
#define B_     64
#define DIM_   512
#define NTOK   320              // H*W = 16*20
#define MTOT   (B_ * NTOK)      // 20480 tokens total
#define HID_   256
#define HEADS_ 8
#define CH_    32               // HID/HEADS
#define NZ_    64

typedef __bf16 bf16_t;
typedef __attribute__((ext_vector_type(16))) __bf16 v16bf;
typedef __attribute__((ext_vector_type(8)))  float  v8f;
typedef __attribute__((ext_vector_type(4)))  unsigned int u32x4_t;
typedef __attribute__((ext_vector_type(8)))  int i32x8_t;
typedef __attribute__((ext_vector_type(4)))  int i32x4_t;

#if __has_builtin(__builtin_amdgcn_tensor_load_to_lds)
#define HAVE_TDM 1
#else
#define HAVE_TDM 0
#endif

__device__ __forceinline__ bf16_t f2bf(float f) {
  union { float f; unsigned u; } x; x.f = f;
  unsigned r = x.u + 0x7FFFu + ((x.u >> 16) & 1u);   // round-to-nearest-even
  union { unsigned short s; bf16_t b; } o;
  o.s = (unsigned short)(r >> 16);
  return o.b;
}

// ---------------------------------------------------------------------------
// (B, C, N) fp32  ->  token-major (B*N, C) bf16 (+ optional fp32 copy)
// ---------------------------------------------------------------------------
__global__ __launch_bounds__(256) void k_transpose(
    const float* __restrict__ x, bf16_t* __restrict__ outH,
    float* __restrict__ outF)
{
  __shared__ float tile[32][33];
  int t = threadIdx.x, tx = t & 31, ty = t >> 5;
  int n0 = blockIdx.x * 32, k0 = blockIdx.y * 32, b = blockIdx.z;
  const float* xp = x + (size_t)b * DIM_ * NTOK;
#pragma unroll
  for (int i = 0; i < 4; i++) {
    int r = ty + i * 8;                                  // channel offset
    tile[r][tx] = xp[(size_t)(k0 + r) * NTOK + n0 + tx]; // coalesced in n
  }
  __syncthreads();
#pragma unroll
  for (int i = 0; i < 4; i++) {
    int r = ty + i * 8;                                  // token offset
    float v = tile[tx][r];                               // channel k0+tx
    size_t o = (size_t)(b * NTOK + n0 + r) * DIM_ + k0 + tx;
    outH[o] = f2bf(v);
    if (outF) outF[o] = v;
  }
}

// Weight (R x C) fp32 row-major -> N-major bf16: out[c*R + r] = in[r*C + c]
__global__ __launch_bounds__(256) void k_wtrans(
    const float* __restrict__ in, bf16_t* __restrict__ out, int R, int C)
{
  __shared__ float tile[32][33];
  int t = threadIdx.x, tx = t & 31, ty = t >> 5;
  int c0 = blockIdx.x * 32, r0 = blockIdx.y * 32;
#pragma unroll
  for (int i = 0; i < 4; i++) {
    int rr = ty + i * 8;
    tile[rr][tx] = in[(size_t)(r0 + rr) * C + c0 + tx];
  }
  __syncthreads();
#pragma unroll
  for (int i = 0; i < 4; i++) {
    int cc = ty + i * 8;
    out[(size_t)(c0 + cc) * R + r0 + tx] = f2bf(tile[tx][cc]);
  }
}

// Plain fp32 -> bf16 convert (Wq/Wk/Wv/Wo are already N-major: out rows = o)
__global__ void k_wconv(const float* __restrict__ in,
                        bf16_t* __restrict__ out, int n)
{
  int i = blockIdx.x * 256 + threadIdx.x;
  if (i < n) out[i] = f2bf(in[i]);
}

// ---------------------------------------------------------------------------
// Tensor Data Mover: DMA one 128-row x 32-col bf16 tile (row stride lda
// elements) into LDS, padding each 16-DWORD row with 4 DWORDs so the LDS
// row stride is 40 halves (80B) -> 16B-aligned rows, conflict-free ds reads.
// D# packing per CDNA5 ISA §8.3/8.4 (group0: count/lds/global/type,
// group1: data_size, pads, dims, strides). Tracked on TENSORcnt.
// ---------------------------------------------------------------------------
#if HAVE_TDM
__device__ __forceinline__ void tdm_load_tile(const bf16_t* gsrc, int lda,
                                              unsigned lds_byte_addr)
{
  unsigned long long ga = (unsigned long long)(uintptr_t)gsrc;
  u32x4_t g0;
  g0.x = 1u;                                            // count=1, user mode
  g0.y = lds_byte_addr;                                 // LDS dest (bytes)
  g0.z = (unsigned)ga;                                  // global_addr[31:0]
  g0.w = (unsigned)((ga >> 32) & 0x01FFFFFFu) | (2u << 30); // [56:32] | type=2
  i32x8_t g1;
  g1[0] = (1 << 16)      // data_size = 2 bytes
        | (1 << 20)      // pad_enable
        | (3 << 22)      // pad_interval: every 16 DWORDs (one 32-elem row)
        | (3 << 25);     // pad_amount: 4 DWORDs  => 40-half LDS row stride
  g1[1] = (int)(32u << 16);    // tensor_dim0 = 32  (bits[79:48], lo16)
  g1[2] = (int)(128u << 16);   // tensor_dim1 = 128 (bits[111:80], lo16)
  g1[3] = (int)(32u << 16);    // tile_dim0  = 32   (bits[127:112])
  g1[4] = 128;                 // tile_dim1  = 128  (bits[143:128])
  g1[5] = lda;                 // tensor_dim0_stride (elements, bits[191:160])
  g1[6] = 0;
  g1[7] = 0;
  i32x4_t gz = {0, 0, 0, 0};
#if defined(__clang_major__) && (__clang_major__ >= 23)
  i32x8_t gz8 = {0, 0, 0, 0, 0, 0, 0, 0};
  __builtin_amdgcn_tensor_load_to_lds(g0, g1, gz, gz, gz8, 0);
#else
  __builtin_amdgcn_tensor_load_to_lds(g0, g1, gz, gz, 0);
#endif
}
#endif

// ---------------------------------------------------------------------------
// Generic bf16 WMMA GEMM: C(MxN) = A(MxK,row-major) * B(KxN, stored N-major)
// + bias + ReLU + up to two fp32 residuals; fp32 and/or bf16 stores.
// Block = 256 threads = 8 waves (wave32), macro-tile 128x128,
// wave tile 32x64 -> 2x4 fragments of v_wmma_f32_16x16x32_bf16.
// A: double-buffered LDS tile filled by the TDM (or b128 copies as fallback);
// B: read straight from global (weights are L2-resident), each lane's 16
// halves (K 8h..8h+7, 16+8h..16+8h+7) = 2x global_load_b128.
// ---------------------------------------------------------------------------
#define SA_LD 40                 // padded LDS row stride in halves (80 B)
__global__ __launch_bounds__(256) void k_gemm(
    int K,
    const bf16_t* __restrict__ A, int lda,
    const bf16_t* __restrict__ Bt,           // Bt[n*K + k]
    const float* __restrict__ bias,
    const float* __restrict__ res0, int ldr0,
    const float* __restrict__ res1, int ldr1,
    int relu,
    float* __restrict__ outF, int ldcF,
    bf16_t* __restrict__ outH, int ldcH)
{
  __shared__ unsigned short sA[2][128 * SA_LD];
  int tid  = threadIdx.x;
  int lane = tid & 31, wid = tid >> 5;
  int wm = wid & 3, wn = wid >> 2;          // 4 (M) x 2 (N) waves
  int half = lane >> 4, l15 = lane & 15;
  int rowBase = blockIdx.y * 128;
  int colBase = blockIdx.x * 128;
  const bf16_t* Atile = A + (size_t)rowBase * lda;

  v8f zero = {0.f, 0.f, 0.f, 0.f, 0.f, 0.f, 0.f, 0.f};
  v8f acc[2][4];
#pragma unroll
  for (int mf = 0; mf < 2; mf++)
#pragma unroll
    for (int nf = 0; nf < 4; nf++) acc[mf][nf] = zero;

#if HAVE_TDM
  // Prologue: wave 0 programs the TDM for tile 0 (generic LDS address low
  // 32 bits are the LDS byte offset per the flat-address aperture rules).
  if (tid < 32)
    tdm_load_tile(Atile, lda, (unsigned)(uintptr_t)&sA[0][0]);
#endif

  int buf = 0;
  for (int k0 = 0; k0 < K; k0 += 32) {
#if HAVE_TDM
    if (tid < 32) __builtin_amdgcn_s_wait_tensorcnt(0);  // tile k0 landed
    __syncthreads();                                     // publish to all waves
    // Overlap: kick off next tile's DMA into the retired buffer while the
    // 8 WMMAs below consume the current one.
    if ((k0 + 32 < K) && (tid < 32))
      tdm_load_tile(Atile + k0 + 32, lda,
                    (unsigned)(uintptr_t)&sA[buf ^ 1][0]);
#else
    // Fallback: vectorized b128 staging (2 loads + 2 LDS stores / thread).
#pragma unroll
    for (int i = 0; i < 2; i++) {
      int q = tid + i * 256;                 // 512 x 16B chunks
      int r = q >> 2, c4 = q & 3;
      uint4 w = *(const uint4*)(Atile + (size_t)r * lda + k0 + c4 * 8);
      *(uint4*)&sA[buf][r * SA_LD + c4 * 8] = w;
    }
    __syncthreads();
#endif
    const unsigned short* sAb = sA[buf];

    // ---- A fragments (ISA 7.12.2: 16-bit A 16x32 layout) ----
    union { v16bf v; unsigned int u[8]; } af[2];
#pragma unroll
    for (int mf = 0; mf < 2; mf++) {
      int mr = wm * 32 + mf * 16 + l15;
#pragma unroll
      for (int j = 0; j < 8; j++) {
        int kk = (j >> 2) * 16 + half * 8 + (j & 3) * 2;
        af[mf].u[j] = *(const unsigned int*)&sAb[mr * SA_LD + kk];
      }
    }

    // ---- B fragments straight from global (N-major => K 8h..8h+7, 16+8h..16+8h+7) ----
#pragma unroll
    for (int nf = 0; nf < 4; nf++) {
      int nc = colBase + wn * 64 + nf * 16 + l15;
      const bf16_t* bp = Bt + (size_t)nc * K + k0 + half * 8;
      union { v16bf v; uint4 q[2]; } bm;
      bm.q[0] = ((const uint4*)bp)[0];
      bm.q[1] = ((const uint4*)bp)[2];
      acc[0][nf] = __builtin_amdgcn_wmma_f32_16x16x32_bf16(
          false, af[0].v, false, bm.v, (short)0, acc[0][nf], false, false);
      acc[1][nf] = __builtin_amdgcn_wmma_f32_16x16x32_bf16(
          false, af[1].v, false, bm.v, (short)0, acc[1][nf], false, false);
    }
    buf ^= 1;
  }

  // ---- fused epilogue (C/D layout: VGPR r -> M = r + 8*half, N = l15) ----
#pragma unroll
  for (int nf = 0; nf < 4; nf++) {
    int col = colBase + wn * 64 + nf * 16 + l15;
    float bv = bias ? bias[col] : 0.f;
#pragma unroll
    for (int mf = 0; mf < 2; mf++) {
#pragma unroll
      for (int r = 0; r < 8; r++) {
        int row = rowBase + wm * 32 + mf * 16 + half * 8 + r;
        float v = acc[mf][nf][r] + bv;
        if (relu) v = v > 0.f ? v : 0.f;
        if (res0) v += res0[(size_t)row * ldr0 + col];
        if (res1) v += res1[(size_t)row * ldr1 + col];
        if (outF) outF[(size_t)row * ldcF + col] = v;
        if (outH) outH[(size_t)row * ldcH + col] = f2bf(v);
      }
    }
  }
}

// ---------------------------------------------------------------------------
// Channel attention per (batch, branch, head). l2-norm over tokens factors
// out into per-channel scales: attn[c][d] = (sum_n q*k) * sq[c]*sk[d]*temp.
// Streams 32-token chunks through LDS; fp32 math; bf16 output.
// ---------------------------------------------------------------------------
__global__ __launch_bounds__(256) void k_attn(
    const float* __restrict__ Qf, const float* __restrict__ Kf,
    const float* __restrict__ Vf, const float* __restrict__ temp,
    bf16_t* __restrict__ outH)
{
  __shared__ float cq[32][33];
  __shared__ float ck[32][33];
  __shared__ float attn[32][33];
  __shared__ float sq[32], sk[32];

  int t = threadIdx.x;
  int bx = blockIdx.x;
  int b = bx >> 4;
  int branch = (bx >> 3) & 1, head = bx & 7;
  int n = branch ? (NTOK - NZ_) : NZ_;
  int tokBase = b * NTOK + (branch ? NZ_ : 0);
  int colBase = head * CH_;

  // per-channel 1/||.||_2 over the token axis
  if (t < 64) {
    int c = t & 31;
    const float* P = (t < 32) ? Qf : Kf;
    float s = 0.f;
    for (int i = 0; i < n; i++) {
      float v = P[(size_t)(tokBase + i) * HID_ + colBase + c];
      s += v * v;
    }
    s = 1.f / fmaxf(sqrtf(s), 1e-12f);
    if (t < 32) sq[c] = s; else sk[c] = s;
  }
  __syncthreads();

  int c = t >> 3, dg = t & 7;
  float accA[4] = {0.f, 0.f, 0.f, 0.f};
  for (int n0 = 0; n0 < n; n0 += 32) {
#pragma unroll
    for (int i = 0; i < 4; i++) {
      int e = t + i * 256;
      int r = e >> 5, cc = e & 31;
      cq[r][cc] = Qf[(size_t)(tokBase + n0 + r) * HID_ + colBase + cc];
      ck[r][cc] = Kf[(size_t)(tokBase + n0 + r) * HID_ + colBase + cc];
    }
    __syncthreads();
#pragma unroll 8
    for (int i = 0; i < 32; i++) {
      float qv = cq[i][c];
#pragma unroll
      for (int j = 0; j < 4; j++) accA[j] += qv * ck[i][dg * 4 + j];
    }
    __syncthreads();
  }
  float tm = temp[head];
#pragma unroll
  for (int j = 0; j < 4; j++)
    attn[c][dg * 4 + j] = accA[j] * sq[c] * sk[dg * 4 + j] * tm;
  __syncthreads();

  if (t < 32) {                       // softmax over d per row c
    float m = -1e30f;
    for (int d = 0; d < 32; d++) m = fmaxf(m, attn[t][d]);
    float s = 0.f;
    for (int d = 0; d < 32; d++) { float e = __expf(attn[t][d] - m); attn[t][d] = e; s += e; }
    float inv = 1.f / s;
    for (int d = 0; d < 32; d++) attn[t][d] *= inv;
  }
  __syncthreads();

  for (int n0 = 0; n0 < n; n0 += 32) {     // out = attn @ v (chunked)
#pragma unroll
    for (int i = 0; i < 4; i++) {
      int e = t + i * 256;
      int r = e >> 5, cc = e & 31;
      cq[r][cc] = Vf[(size_t)(tokBase + n0 + r) * HID_ + colBase + cc];
    }
    __syncthreads();
    int ir = t >> 3, cg = t & 7;
#pragma unroll
    for (int j = 0; j < 4; j++) {
      int co = cg * 4 + j;
      float o = 0.f;
#pragma unroll 8
      for (int d = 0; d < 32; d++) o += attn[co][d] * cq[ir][d];
      outH[(size_t)(tokBase + n0 + ir) * HID_ + colBase + co] = f2bf(o);
    }
    __syncthreads();
  }
}

// ---------------------------------------------------------------------------
// LayerNorm over DIM + transpose back to (B, C, H, W). 16 token-rows / block.
// ---------------------------------------------------------------------------
__global__ __launch_bounds__(256) void k_ln(
    const float* __restrict__ pre, const float* __restrict__ gamma,
    const float* __restrict__ beta, float* __restrict__ out)
{
  __shared__ float sp[16][513];
  __shared__ float redS[16][16];
  __shared__ float redQ[16][16];
  __shared__ float muS[16], rsS[16];
  int t = threadIdx.x;
  int b = blockIdx.y, n0 = blockIdx.x * 16;
#pragma unroll
  for (int i = 0; i < 32; i++) {
    int e = t + i * 256;
    int row = e >> 9, col = e & 511;
    sp[row][col] = pre[(size_t)(b * NTOK + n0 + row) * DIM_ + col];
  }
  __syncthreads();
  {
    int row = t >> 4, seg = t & 15;
    float s = 0.f, q = 0.f;
#pragma unroll
    for (int c2 = 0; c2 < 32; c2++) {
      float v = sp[row][seg * 32 + c2];
      s += v; q += v * v;
    }
    redS[row][seg] = s; redQ[row][seg] = q;
  }
  __syncthreads();
  if (t < 16) {
    float s = 0.f, q = 0.f;
    for (int i = 0; i < 16; i++) { s += redS[t][i]; q += redQ[t][i]; }
    float mu = s * (1.f / 512.f);
    float var = q * (1.f / 512.f) - mu * mu;
    muS[t] = mu;
    rsS[t] = rsqrtf(var + 1e-5f);
  }
  __syncthreads();
  float* op = out + (size_t)b * DIM_ * NTOK;
#pragma unroll
  for (int i = 0; i < 32; i++) {
    int e = t + i * 256;
    int col = e >> 4, row = e & 15;              // 16 consecutive tokens/col
    float v = (sp[row][col] - muS[row]) * rsS[row] * gamma[col] + beta[col];
    op[(size_t)col * NTOK + n0 + row] = v;
  }
}

// ---------------------------------------------------------------------------
// Host orchestration
// ---------------------------------------------------------------------------
extern "C" void kernel_launch(void* const* d_in, const int* in_sizes, int n_in,
                              void* d_out, int out_size, void* d_ws, size_t ws_size,
                              hipStream_t stream)
{
  (void)in_sizes; (void)n_in; (void)out_size; (void)ws_size;
  const float* x1     = (const float*)d_in[0];
  const float* x2     = (const float*)d_in[1];
  const float* W_lin  = (const float*)d_in[2];
  const float* b_lin  = (const float*)d_in[3];
  const float* W_down = (const float*)d_in[4];
  const float* b_down = (const float*)d_in[5];
  const float* W_up   = (const float*)d_in[6];
  const float* b_up   = (const float*)d_in[7];
  const float* Wq     = (const float*)d_in[8];
  const float* Wk     = (const float*)d_in[9];
  const float* Wv     = (const float*)d_in[10];
  const float* Wo     = (const float*)d_in[11];
  const float* temp   = (const float*)d_in[12];
  const float* W_end  = (const float*)d_in[13];
  const float* b_end  = (const float*)d_in[14];
  const float* gamma  = (const float*)d_in[15];
  const float* beta   = (const float*)d_in[16];
  float* out = (float*)d_out;

  char* p = (char*)d_ws;
  auto carve = [&](size_t bytes) -> char* {
    char* r = p; p += (bytes + 255) & ~(size_t)255; return r;
  };
  bf16_t* t1h   = (bf16_t*)carve((size_t)MTOT * DIM_ * 2);
  float*  t1f   = (float*) carve((size_t)MTOT * DIM_ * 4);
  bf16_t* t2h   = (bf16_t*)carve((size_t)MTOT * DIM_ * 2);
  bf16_t* WlinT = (bf16_t*)carve((size_t)512 * 1024 * 2);
  bf16_t* WdownT= (bf16_t*)carve((size_t)512 * 256 * 2);
  bf16_t* WupT  = (bf16_t*)carve((size_t)256 * 512 * 2);
  bf16_t* WendT = (bf16_t*)carve((size_t)512 * 512 * 2);
  bf16_t* Wqb   = (bf16_t*)carve((size_t)256 * 256 * 2);
  bf16_t* Wkb   = (bf16_t*)carve((size_t)256 * 256 * 2);
  bf16_t* Wvb   = (bf16_t*)carve((size_t)256 * 256 * 2);
  bf16_t* Wob   = (bf16_t*)carve((size_t)256 * 256 * 2);
  float*  h1f   = (float*) carve((size_t)MTOT * 1024 * 4);
  bf16_t* h1h   = (bf16_t*)carve((size_t)MTOT * 1024 * 2);
  bf16_t* u2h   = (bf16_t*)carve((size_t)MTOT * 512 * 2);
  float*  af32  = (float*) carve((size_t)MTOT * 256 * 4);
  bf16_t* ah    = (bf16_t*)carve((size_t)MTOT * 256 * 2);
  bf16_t* bh    = (bf16_t*)carve((size_t)MTOT * 256 * 2);
  float*  Qf    = (float*) carve((size_t)MTOT * 256 * 4);
  float*  Kfp   = (float*) carve((size_t)MTOT * 256 * 4);
  float*  Vfp   = (float*) carve((size_t)MTOT * 256 * 4);
  bf16_t* aoH   = (bf16_t*)carve((size_t)MTOT * 256 * 2);
  bf16_t* tokH  = (bf16_t*)carve((size_t)MTOT * 256 * 2);
  bf16_t* yH    = (bf16_t*)carve((size_t)MTOT * 512 * 2);
  float*  preF  = (float*) carve((size_t)MTOT * 512 * 4);

  dim3 blk(256);

  // input transposes + weight conversions (to N-major bf16)
  k_transpose<<<dim3(10, 16, 64), blk, 0, stream>>>(x1, t1h, t1f);
  k_transpose<<<dim3(10, 16, 64), blk, 0, stream>>>(x2, t2h, nullptr);
  k_wtrans<<<dim3(32, 16), blk, 0, stream>>>(W_lin,  WlinT,  512, 1024);
  k_wtrans<<<dim3(8, 16),  blk, 0, stream>>>(W_down, WdownT, 512, 256);
  k_wtrans<<<dim3(16, 8),  blk, 0, stream>>>(W_up,   WupT,   256, 512);
  k_wtrans<<<dim3(16, 16), blk, 0, stream>>>(W_end,  WendT,  512, 512);
  k_wconv<<<256, blk, 0, stream>>>(Wq, Wqb, 256 * 256);
  k_wconv<<<256, blk, 0, stream>>>(Wk, Wkb, 256 * 256);
  k_wconv<<<256, blk, 0, stream>>>(Wv, Wvb, 256 * 256);
  k_wconv<<<256, blk, 0, stream>>>(Wo, Wob, 256 * 256);

  // h1 = relu(t1 @ W_lin + b_lin): keep fp32 (y1,u1 residuals) + bf16
  k_gemm<<<dim3(8, MTOT / 128), blk, 0, stream>>>(
      512, t1h, 512, WlinT, b_lin, nullptr, 0, nullptr, 0, 1,
      h1f, 1024, h1h, 1024);
  // u2 = relu(t2 @ W_lin[:,512:] + b_lin[512:]) (y2 is never used)
  k_gemm<<<dim3(4, MTOT / 128), blk, 0, stream>>>(
      512, t2h, 512, WlinT + (size_t)512 * 512, b_lin + 512,
      nullptr, 0, nullptr, 0, 1, nullptr, 0, u2h, 512);
  // a  = relu(u1 @ W_down + b_down): fp32 (res0) + bf16
  k_gemm<<<dim3(2, MTOT / 128), blk, 0, stream>>>(
      512, h1h + 512, 1024, WdownT, b_down, nullptr, 0, nullptr, 0, 1,
      af32, 256, ah, 256);
  // b_ = relu(u2 @ W_down + b_down): bf16 only
  k_gemm<<<dim3(2, MTOT / 128), blk, 0, stream>>>(
      512, u2h, 512, WdownT, b_down, nullptr, 0, nullptr, 0, 1,
      nullptr, 0, bh, 256);
  // Q = b_ @ Wq^T,  K = a @ Wk^T,  V = a @ Wv^T   (fp32 for attention math)
  k_gemm<<<dim3(2, MTOT / 128), blk, 0, stream>>>(
      256, bh, 256, Wqb, nullptr, nullptr, 0, nullptr, 0, 0, Qf, 256, nullptr, 0);
  k_gemm<<<dim3(2, MTOT / 128), blk, 0, stream>>>(
      256, ah, 256, Wkb, nullptr, nullptr, 0, nullptr, 0, 0, Kfp, 256, nullptr, 0);
  k_gemm<<<dim3(2, MTOT / 128), blk, 0, stream>>>(
      256, ah, 256, Wvb, nullptr, nullptr, 0, nullptr, 0, 0, Vfp, 256, nullptr, 0);
  // channel attention per (b, branch, head)
  k_attn<<<dim3(B_ * 2 * HEADS_), blk, 0, stream>>>(Qf, Kfp, Vfp, temp, aoH);
  // tok = a + attn_out @ Wo^T
  k_gemm<<<dim3(2, MTOT / 128), blk, 0, stream>>>(
      256, aoH, 256, Wob, nullptr, af32, 256, nullptr, 0, 0,
      nullptr, 0, tokH, 256);
  // y = y1 + u1 + tok @ W_up + b_up
  k_gemm<<<dim3(4, MTOT / 128), blk, 0, stream>>>(
      256, tokH, 256, WupT, b_up, h1f, 1024, h1f + 512, 1024, 0,
      nullptr, 0, yH, 512);
  // pre = t1 + y @ W_end + b_end
  k_gemm<<<dim3(4, MTOT / 128), blk, 0, stream>>>(
      512, yH, 512, WendT, b_end, t1f, 512, nullptr, 0, 0,
      preF, 512, nullptr, 0);
  // LayerNorm + transpose back to (B, C, H, W)
  k_ln<<<dim3(NTOK / 16, B_), blk, 0, stream>>>(preF, gamma, beta, out);
}
